// CWS_10952166605290
// MI455X (gfx1250) — matrix operand order repaired and checked by reading, not compiled, not run on hardware
//
#include <hip/hip_runtime.h>
#include <math.h>

typedef __attribute__((ext_vector_type(16))) _Float16 v16h;
typedef __attribute__((ext_vector_type(8)))  _Float16 v8h;
typedef __attribute__((ext_vector_type(8)))  float    v8f;
typedef __attribute__((ext_vector_type(4)))  unsigned int u32x4;
typedef __attribute__((ext_vector_type(8)))  int      i32x8;
typedef __attribute__((ext_vector_type(4)))  int      i32x4;

#define B_  64
#define S_  512
#define E_  256
#define HH_ 128
#define G_  512   // 4*HH
#define T_  4
#define HP_ 136   // padded f16 row stride (272B) -> distinct LDS bank quads per lane

#if __has_builtin(__builtin_amdgcn_tensor_load_to_lds)
#define HAVE_TDM 1
#else
#define HAVE_TDM 0
#endif

// --------------------------------------------------------------------------
// Kernel 1: fused embedding gather + input projection GEMM (per direction)
//   Xp[(b*S+s)*G + g] = sum_e embed[sentence[b,s], e] * W_ih[g, e] + b_ih[g] + b_hh[g]
// M = B*S = 32768, K = 256, N = 512.  One 16x16 tile per wave, 8x WMMA K-loop.
// Frags assembled from contiguous float4 (b128) loads per ISA layout:
//   A and B: two 8-elem k-runs at kb+8*half and kb+16+8*half.
// --------------------------------------------------------------------------
__global__ __launch_bounds__(256) void k_input_proj(
    const int* __restrict__ sentence,
    const float* __restrict__ embed,
    const float* __restrict__ Wih_f, const float* __restrict__ bih_f, const float* __restrict__ bhh_f,
    const float* __restrict__ Wih_b, const float* __restrict__ bih_b, const float* __restrict__ bhh_b,
    float* __restrict__ Xp_f, float* __restrict__ Xp_b)
{
  const int lane = threadIdx.x & 31;
  const int wave = threadIdx.x >> 5;
  const int hl   = lane >> 4;
  const int l16  = lane & 15;

  const int NT = G_ / 16;           // 32
  const int MT = (B_ * S_) / 16;    // 2048
  int tile = blockIdx.x * 8 + wave;
  int dir  = tile / (MT * NT);
  int rem  = tile % (MT * NT);
  int mt   = rem / NT;
  int nt   = rem % NT;

  const float* Wih = dir ? Wih_b : Wih_f;
  const float* bih = dir ? bih_b : bih_f;
  const float* bhh = dir ? bhh_b : bhh_f;
  float*       Xp  = dir ? Xp_b  : Xp_f;

  int m   = mt * 16 + l16;                       // flat (b*S + s)
  int tok = sentence[m];
  const float* arow = embed + (long)tok * E_;
  int g = nt * 16 + l16;
  const float* brow = Wih + (long)g * E_;

  v8f acc = {};
  #pragma unroll 1
  for (int kb = 0; kb < E_; kb += 32) {
    const float* ap = arow + kb + (hl << 3);
    float4 fa0 = *(const float4*)(ap);
    float4 fa1 = *(const float4*)(ap + 4);
    float4 fa2 = *(const float4*)(ap + 16);
    float4 fa3 = *(const float4*)(ap + 20);
    const float* bp = brow + kb + (hl << 3);
    float4 fb0 = *(const float4*)(bp);
    float4 fb1 = *(const float4*)(bp + 4);
    float4 fb2 = *(const float4*)(bp + 16);
    float4 fb3 = *(const float4*)(bp + 20);
    v16h a, bf;
    a[0]=(_Float16)fa0.x;  a[1]=(_Float16)fa0.y;  a[2]=(_Float16)fa0.z;  a[3]=(_Float16)fa0.w;
    a[4]=(_Float16)fa1.x;  a[5]=(_Float16)fa1.y;  a[6]=(_Float16)fa1.z;  a[7]=(_Float16)fa1.w;
    a[8]=(_Float16)fa2.x;  a[9]=(_Float16)fa2.y;  a[10]=(_Float16)fa2.z; a[11]=(_Float16)fa2.w;
    a[12]=(_Float16)fa3.x; a[13]=(_Float16)fa3.y; a[14]=(_Float16)fa3.z; a[15]=(_Float16)fa3.w;
    bf[0]=(_Float16)fb0.x;  bf[1]=(_Float16)fb0.y;  bf[2]=(_Float16)fb0.z;  bf[3]=(_Float16)fb0.w;
    bf[4]=(_Float16)fb1.x;  bf[5]=(_Float16)fb1.y;  bf[6]=(_Float16)fb1.z;  bf[7]=(_Float16)fb1.w;
    bf[8]=(_Float16)fb2.x;  bf[9]=(_Float16)fb2.y;  bf[10]=(_Float16)fb2.z; bf[11]=(_Float16)fb2.w;
    bf[12]=(_Float16)fb3.x; bf[13]=(_Float16)fb3.y; bf[14]=(_Float16)fb3.z; bf[15]=(_Float16)fb3.w;
    acc = __builtin_amdgcn_wmma_f32_16x16x32_f16(false, a, false, bf, (short)0, acc, false, false);
  }

  float bias = bih[g] + bhh[g];
  #pragma unroll
  for (int r = 0; r < 8; ++r) {
    int row = mt * 16 + hl * 8 + r;      // C/D: M = r + 8*half, N = l16
    Xp[(long)row * G_ + g] = acc[r] + bias;
  }
}

// --------------------------------------------------------------------------
// Kernel 2: persistent LSTM scan.  8 workgroups = {2 dirs} x {4 batch groups
// of 16 rows} -> 8 WGPs busy (batch rows are independent recurrences).
// LDS layout (dynamic, no static shared => dynamic base offset 0):
//   [0)       Whh  f16, padded stride HP_=136 : 512*136*2 = 139264 B
//   [139264)  h    f16, padded stride 136     : 16*136*2  =   4352 B
//   [143616)  c    f32                         : 16*128*4  =   8192 B
//   [151808)  gates f32                        : 16*512*4  =  32768 B
//   [184576)  Xp slab double buffer f32        : 2*16*512*4=  65536 B
// total 250112 B (< 320KB/WGP).  Per step: TDM DMA of next step's Xp slab
// overlapped with 32 WMMA tiles (2 per wave), then masked LSTM cell.
// --------------------------------------------------------------------------
#define LDS_WHH  0
#define LDS_H    139264
#define LDS_C    143616
#define LDS_G    151808
#define LDS_XP   184576
#define LDS_TOT  250112

__global__ __launch_bounds__(512) void k_lstm_scan(
    const float* __restrict__ Whh_f, const float* __restrict__ Whh_b,
    const float* __restrict__ h0, const float* __restrict__ c0,
    const int* __restrict__ length,
    const float* __restrict__ Xp_f, const float* __restrict__ Xp_b,
    float* __restrict__ hf_out, float* __restrict__ hb_out)
{
  extern __shared__ char smem[];
  _Float16* Whh_s = (_Float16*)(smem + LDS_WHH);
  _Float16* h_s   = (_Float16*)(smem + LDS_H);
  float*    c_s   = (float*)(smem + LDS_C);
  float*    g_s   = (float*)(smem + LDS_G);
  float*    xp_s  = (float*)(smem + LDS_XP);

  const int dir = blockIdx.x >> 2;       // 0 fwd, 1 bwd
  const int bg  = blockIdx.x & 3;        // batch group: rows [bg*16, bg*16+16)
  const int b0  = bg * 16;
  const float* Whh = dir ? Whh_b : Whh_f;
  const float* Xp  = dir ? Xp_b  : Xp_f;
  float* hout      = dir ? hb_out : hf_out;

  const int tid  = threadIdx.x;
  const int lane = tid & 31;
  const int wave = tid >> 5;
  const int hl   = lane >> 4;
  const int l16  = lane & 15;

  // stage W_hh (f16, padded rows) + initial h/c
  for (int i = tid; i < G_ * HH_; i += 512)
    Whh_s[(i >> 7) * HP_ + (i & 127)] = (_Float16)Whh[i];
  for (int e = tid; e < 16 * HH_; e += 512) {
    int lb = e >> 7, j = e & 127;
    h_s[lb * HP_ + j] = (_Float16)h0[dir * B_ * HH_ + (b0 + lb) * HH_ + j];
    c_s[e]            = c0[dir * B_ * HH_ + (b0 + lb) * HH_ + j];
  }

#if HAVE_TDM
  // TDM descriptor issue: 2D tile, 16 rows x 512 f32, row stride S_*G_ elems.
  auto tdm_issue = [&](int t, int buf) {
    unsigned long long ga = (unsigned long long)Xp
                          + (((unsigned long long)b0 * S_ + (unsigned)t) * G_) * 4ull;
    unsigned ldsoff = (unsigned)(LDS_XP + buf * 16 * G_ * 4);
    u32x4 d0; i32x8 d1; i32x4 d2 = {0, 0, 0, 0}; i32x4 d3 = {0, 0, 0, 0};
    i32x8 d4 = {0, 0, 0, 0, 0, 0, 0, 0};
    d0[0] = 1u;                                   // count=1, user mode
    d0[1] = ldsoff;                               // lds_addr
    d0[2] = (unsigned)(ga & 0xFFFFFFFFull);       // global_addr[31:0]
    d0[3] = (unsigned)((ga >> 32) & 0x01FFFFFFu) | (2u << 30);   // addr[56:32] | type=2
    d1[0] = 2 << 16;                              // data_size = 4B
    d1[1] = (int)(512u << 16);                    // tensor_dim0 = 512 (bits 79:48 lo)
    d1[2] = (int)(16u << 16);                     // tensor_dim1 = 16  (bits 111:80 lo)
    d1[3] = (int)(512u << 16);                    // tile_dim0 = 512   (bits 127:112)
    d1[4] = 16;                                   // tile_dim1 = 16
    d1[5] = (int)(S_ * G_);                       // tensor_dim0_stride = 262144
    d1[6] = 0; d1[7] = 0;
    // 6-arg clang-23 form: (g0, g1, g2, g3, g_extra, cpol)
    __builtin_amdgcn_tensor_load_to_lds(d0, d1, d2, d3, d4, 0);
  };
  if (wave == 0) tdm_issue(dir ? S_ - 1 : 0, 0);
#endif
  __syncthreads();

  int cur = 0;
  for (int step = 0; step < S_; ++step) {
    int t = dir ? (S_ - 1 - step) : step;

#if HAVE_TDM
    if (wave == 0) {
      __builtin_amdgcn_s_wait_tensorcnt(0);       // slab(cur) landed in LDS
    }
    __syncthreads();                              // publish to all waves
    if (wave == 0 && step + 1 < S_)
      tdm_issue(dir ? t - 1 : t + 1, cur ^ 1);    // overlap DMA with compute
#endif

    // ---- recurrent GEMM: gates = h @ Whh^T + Xp slab; 2 tiles per wave
    #pragma unroll 1
    for (int i = 0; i < 2; ++i) {
      int nt = (wave << 1) | i;
      int g  = nt * 16 + l16;
      v8f acc = {};
      #pragma unroll
      for (int kc = 0; kc < HH_; kc += 32) {
        const _Float16* ap = h_s + l16 * HP_ + kc + (hl << 3);
        v8h alo = *(const v8h*)(ap);
        v8h ahi = *(const v8h*)(ap + 16);
        const _Float16* bp = Whh_s + g * HP_ + kc + (hl << 3);
        v8h blo = *(const v8h*)(bp);
        v8h bhi = *(const v8h*)(bp + 16);
        v16h a  = __builtin_shufflevector(alo, ahi, 0,1,2,3,4,5,6,7,8,9,10,11,12,13,14,15);
        v16h bf = __builtin_shufflevector(blo, bhi, 0,1,2,3,4,5,6,7,8,9,10,11,12,13,14,15);
        acc = __builtin_amdgcn_wmma_f32_16x16x32_f16(false, a, false, bf, (short)0, acc, false, false);
      }
      #pragma unroll
      for (int r = 0; r < 8; ++r) {
        int lr = hl * 8 + r;                     // local batch row 0..15
#if HAVE_TDM
        float xv = xp_s[cur * (16 * G_) + lr * G_ + g];
#else
        float xv = Xp[((long)(b0 + lr) * S_ + t) * G_ + g];
#endif
        g_s[lr * G_ + g] = acc[r] + xv;
      }
    }
    __syncthreads();

    // ---- elementwise LSTM cell + length mask (4 elems/thread)
    for (int e = tid; e < 16 * HH_; e += 512) {
      int lb = e >> 7, j = e & 127;
      int b  = b0 + lb;
      float ig = g_s[lb * G_ + j];
      float fg = g_s[lb * G_ + 128 + j];
      float gg = g_s[lb * G_ + 256 + j];
      float og = g_s[lb * G_ + 384 + j];
      ig = 1.f / (1.f + __expf(-ig));
      fg = 1.f / (1.f + __expf(-fg));
      gg = tanhf(gg);
      og = 1.f / (1.f + __expf(-og));
      float cn = fg * c_s[e] + ig * gg;
      float hn = og * tanhf(cn);
      float hv;
      if (t < length[b]) {
        c_s[e] = cn;
        h_s[lb * HP_ + j] = (_Float16)hn;
        hv = hn;
      } else {
        hv = (float)h_s[lb * HP_ + j];
      }
      hout[((long)b * S_ + t) * HH_ + j] = hv;
    }
    __syncthreads();
    cur ^= 1;
  }
}

// --------------------------------------------------------------------------
// Kernel 3: emissions = [hf|hb] @ W_out^T + b_out (lstm_out zeroed by mask).
// --------------------------------------------------------------------------
__global__ __launch_bounds__(256) void k_emissions(
    const float* __restrict__ hf, const float* __restrict__ hb,
    const float* __restrict__ Wout, const float* __restrict__ bout,
    const int* __restrict__ length,
    float* __restrict__ emis)
{
  int idx = blockIdx.x * 256 + threadIdx.x;   // b*S*T + s*T + tag
  if (idx >= B_ * S_ * T_) return;
  int tag = idx & 3;
  int s   = (idx >> 2) & (S_ - 1);
  int b   = idx >> 11;
  float acc = bout[tag];
  if (s < length[b]) {
    const float* w  = Wout + tag * (2 * HH_);
    const float* pf = hf + ((long)b * S_ + s) * HH_;
    const float* pb = hb + ((long)b * S_ + s) * HH_;
    float a = 0.f;
    #pragma unroll 4
    for (int j = 0; j < HH_; ++j) a += pf[j] * w[j] + pb[j] * w[HH_ + j];
    acc += a;
  }
  emis[idx] = acc;
}

// --------------------------------------------------------------------------
// Kernel 4: CRF numerator + forward algorithm, one thread per sequence.
// --------------------------------------------------------------------------
__global__ __launch_bounds__(64) void k_crf(
    const int* __restrict__ tags, const int* __restrict__ length,
    const float* __restrict__ start_trans, const float* __restrict__ end_trans,
    const float* __restrict__ trans, const float* __restrict__ emis,
    float* __restrict__ out)
{
  __shared__ float red[64];
  int b = threadIdx.x;
  const int*   tg = tags + b * S_;
  const float* em = emis + (long)b * S_ * T_;
  int len = length[b];

  float tr[16], st[4], et[4];
  #pragma unroll
  for (int i = 0; i < 16; ++i) tr[i] = trans[i];
  #pragma unroll
  for (int i = 0; i < 4; ++i) { st[i] = start_trans[i]; et[i] = end_trans[i]; }

  float num = st[tg[0]] + em[tg[0]];
  for (int t = 1; t < S_; ++t) {
    if (t >= len) break;               // mask monotone per row
    num += tr[tg[t - 1] * 4 + tg[t]] + em[t * 4 + tg[t]];
  }
  num += et[tg[len - 1]];

  float alpha[4];
  #pragma unroll
  for (int j = 0; j < 4; ++j) alpha[j] = st[j] + em[j];
  for (int t = 1; t < S_; ++t) {
    if (t >= len) break;
    float na[4];
    #pragma unroll
    for (int j = 0; j < 4; ++j) {
      float v0 = alpha[0] + tr[0 * 4 + j];
      float v1 = alpha[1] + tr[1 * 4 + j];
      float v2 = alpha[2] + tr[2 * 4 + j];
      float v3 = alpha[3] + tr[3 * 4 + j];
      float mx = fmaxf(fmaxf(v0, v1), fmaxf(v2, v3));
      float sum = expf(v0 - mx) + expf(v1 - mx) + expf(v2 - mx) + expf(v3 - mx);
      na[j] = mx + logf(sum) + em[t * 4 + j];
    }
    #pragma unroll
    for (int j = 0; j < 4; ++j) alpha[j] = na[j];
  }
  float z0 = alpha[0] + et[0], z1 = alpha[1] + et[1];
  float z2 = alpha[2] + et[2], z3 = alpha[3] + et[3];
  float mx = fmaxf(fmaxf(z0, z1), fmaxf(z2, z3));
  float logZ = mx + logf(expf(z0 - mx) + expf(z1 - mx) + expf(z2 - mx) + expf(z3 - mx));

  red[b] = num - logZ;
  __syncthreads();
  if (b == 0) {
    float s = 0.f;
    #pragma unroll
    for (int i = 0; i < 64; ++i) s += red[i];
    out[0] = -s / 64.f;
  }
}

// --------------------------------------------------------------------------
extern "C" void kernel_launch(void* const* d_in, const int* in_sizes, int n_in,
                              void* d_out, int out_size, void* d_ws, size_t ws_size,
                              hipStream_t stream) {
  const int*   sentence    = (const int*)d_in[0];
  const int*   tags        = (const int*)d_in[1];
  // d_in[2] = mask: reconstructed from length (bool dtype ambiguity)
  const int*   length      = (const int*)d_in[3];
  const float* embed       = (const float*)d_in[4];
  const float* Wih_f       = (const float*)d_in[5];
  const float* Whh_f       = (const float*)d_in[6];
  const float* bih_f       = (const float*)d_in[7];
  const float* bhh_f       = (const float*)d_in[8];
  const float* Wih_b       = (const float*)d_in[9];
  const float* Whh_b       = (const float*)d_in[10];
  const float* bih_b       = (const float*)d_in[11];
  const float* bhh_b       = (const float*)d_in[12];
  const float* Wout        = (const float*)d_in[13];
  const float* bout        = (const float*)d_in[14];
  const float* start_trans = (const float*)d_in[15];
  const float* end_trans   = (const float*)d_in[16];
  const float* trans       = (const float*)d_in[17];
  const float* h0          = (const float*)d_in[18];
  const float* c0          = (const float*)d_in[19];

  float* ws   = (float*)d_ws;
  float* Xp_f = ws;
  float* Xp_b = Xp_f + (size_t)B_ * S_ * G_;
  float* hf   = Xp_b + (size_t)B_ * S_ * G_;
  float* hb   = hf + (size_t)B_ * S_ * HH_;
  float* emis = hb + (size_t)B_ * S_ * HH_;

  k_input_proj<<<16384, 256, 0, stream>>>(sentence, embed,
                                          Wih_f, bih_f, bhh_f,
                                          Wih_b, bih_b, bhh_b,
                                          Xp_f, Xp_b);
  // 8 persistent workgroups (2 dirs x 4 batch groups), 250112B dynamic LDS each
  k_lstm_scan<<<8, 512, LDS_TOT, stream>>>(
      Whh_f, Whh_b, h0, c0, length, Xp_f, Xp_b, hf, hb);
  k_emissions<<<512, 256, 0, stream>>>(hf, hb, Wout, bout, length, emis);
  k_crf<<<1, 64, 0, stream>>>(tags, length, start_trans, end_trans, trans, emis,
                              (float*)d_out);
}
